// GroupedQueryAttention_24567212933487
// MI455X (gfx1250) — matrix operand order repaired and checked by reading, not compiled, not run on hardware
//
#include <hip/hip_runtime.h>

// Problem constants (match reference)
#define BB   2
#define NN   2048
#define DD   2048
#define HH   32
#define KVH  8
#define HD   64
#define GG   (HH / KVH)   // 4
#define MM   (BB * NN)    // 4096 rows for all GEMMs

typedef unsigned short u16;
typedef __attribute__((ext_vector_type(16))) __bf16 bf16x16;
typedef __attribute__((ext_vector_type(8)))  float  f32x8;
typedef unsigned int u32x4 __attribute__((ext_vector_type(4)));
typedef int          i32x8 __attribute__((ext_vector_type(8)));
typedef int          i32x4 __attribute__((ext_vector_type(4)));

union FragU { bf16x16 v; uint4 q[2]; };

__device__ __forceinline__ u16 f32_to_bf16(float f) {
  union { float f; unsigned u; } v; v.f = f;
  unsigned u = v.u;
  unsigned r = u + 0x7FFFu + ((u >> 16) & 1u);   // round-to-nearest-even
  return (u16)(r >> 16);
}

// LDS byte offset of a __shared__ pointer (flat addr low dword == LDS offset)
__device__ __forceinline__ unsigned lds_off(const void* p) {
  return (unsigned)(size_t)p;
}

// ---------------------------------------------------------------- TDM 2D tile load
// Loads tile_d1 rows of tile_d0 bf16 elements (row stride stride0 elements in
// global memory) into LDS at lds_addr, inserting pad_amount dwords of LDS
// padding every 2^(pad_interval+1) dwords streamed (ISA 08_async_tensor §8).
// pad codes used here: interval=4 (32 dwords = 128B), amount=7 (8 dwords = 32B)
// -> LDS row pitch 160B = 80 elements.
__device__ __forceinline__ void tdm_load_2d(unsigned lds_addr, const void* gptr,
                                            unsigned tile_d0, unsigned tile_d1,
                                            unsigned stride0_elems,
                                            unsigned pad_interval, unsigned pad_amount) {
  unsigned long long ga = (unsigned long long)(size_t)gptr;
  const unsigned td0 = 0x40000000u;   // huge tensor dims: tile never clamps
  const unsigned td1 = 0x40000000u;
  u32x4 g0;
  g0[0] = 1u;                                                // count=1 (valid user D#)
  g0[1] = lds_addr;                                          // lds_addr (bytes)
  g0[2] = (unsigned)(ga & 0xFFFFFFFFull);                    // global_addr[31:0]
  g0[3] = (unsigned)((ga >> 32) & 0x01FFFFFFull) | (2u << 30); // global_addr[56:32] | type=2
  i32x8 g1;
  g1[0] = (int)((1u << 16) |                 // data_size = 1 -> 2 bytes
                (1u << 20) |                 // pad_enable
                (pad_interval << 22) |
                (pad_amount << 25));
  g1[1] = (int)((td0 & 0xFFFFu) << 16);                      // tensor_dim0[15:0]
  g1[2] = (int)((td0 >> 16) | ((td1 & 0xFFFFu) << 16));      // dim0 hi | dim1 lo
  g1[3] = (int)((td1 >> 16) | (tile_d0 << 16));              // dim1 hi | tile_dim0
  g1[4] = (int)(tile_d1);                                    // tile_dim1 (tile_dim2=0)
  g1[5] = (int)stride0_elems;                                // tensor_dim0_stride[31:0]
  g1[6] = 0;                                                 // stride0 hi | stride1 lo
  g1[7] = 0;
  i32x4 z4 = {0, 0, 0, 0};
  i32x8 z8 = {0, 0, 0, 0, 0, 0, 0, 0};
  // 6-arg form (clang-23 / therock-10.0 signature)
  __builtin_amdgcn_tensor_load_to_lds(g0, g1, z4, z4, z8, 0);
}

// A-operand fragment (16x32 bf16): lane half h holds K = {h*8..h*8+7} U {16+h*8..16+h*8+7}
__device__ __forceinline__ bf16x16 frag_a(const u16* tile, int ld) {
  int l = threadIdx.x & 31;
  const u16* p = tile + (l & 15) * ld + (l >> 4) * 8;
  FragU f;
  f.q[0] = *(const uint4*)(p);
  f.q[1] = *(const uint4*)(p + 16);
  return f.v;
}

// B-operand fragment (32x16 bf16), source stored K-contiguous per output column n:
// lane half h holds K = {h*8..h*8+7} U {16+h*8..16+h*8+7} of column n = lane&15
__device__ __forceinline__ bf16x16 frag_b(const u16* tile, int ld) {
  int l = threadIdx.x & 31;
  const u16* p = tile + (l & 15) * ld + (l >> 4) * 8;
  FragU f;
  f.q[0] = *(const uint4*)(p);
  f.q[1] = *(const uint4*)(p + 16);
  return f.v;
}

// ---------------------------------------------------------------- prep kernels
__global__ void cvt_bf16_kernel(const float* __restrict__ src, u16* __restrict__ dst, size_t n) {
  size_t i = blockIdx.x * (size_t)blockDim.x + threadIdx.x;
  if (i < n) dst[i] = f32_to_bf16(src[i]);
}

// W (K x Nc, f32 row-major) -> Wt (Nc x K, bf16 row-major)
__global__ void transpose_cvt_kernel(const float* __restrict__ W, u16* __restrict__ Wt, int K, int Nc) {
  size_t i = blockIdx.x * (size_t)blockDim.x + threadIdx.x;
  size_t total = (size_t)K * Nc;
  if (i >= total) return;
  int n = (int)(i % Nc);
  int k = (int)(i / Nc);
  Wt[(size_t)n * K + k] = f32_to_bf16(W[i]);
}

// src (b*npos+pos, heads*HD) f32  ->  dst ((b*heads+h)*npos+pos, HD) bf16, optional interleaved RoPE
__global__ void rope_cvt_kernel(const float* __restrict__ src, u16* __restrict__ dst,
                                const float* __restrict__ fc, const float* __restrict__ fs,
                                int heads, int do_rope) {
  size_t idx = blockIdx.x * (size_t)blockDim.x + threadIdx.x;
  size_t total = (size_t)BB * heads * NN * (HD / 2);
  if (idx >= total) return;
  int i = (int)(idx & (HD / 2 - 1));        // pair index
  size_t t = idx >> 5;                       // HD/2 == 32
  int pos = (int)(t % NN); t /= NN;
  int h   = (int)(t % heads); t /= heads;
  int b   = (int)t;
  const float* s = src + ((size_t)b * NN + pos) * (heads * HD) + h * HD + i * 2;
  float xr = s[0], xi = s[1];
  float o0 = xr, o1 = xi;
  if (do_rope) {
    float c = fc[pos * (HD / 2) + i];
    float sn = fs[pos * (HD / 2) + i];
    o0 = xr * c - xi * sn;
    o1 = xr * sn + xi * c;
  }
  u16* d = dst + (((size_t)b * heads + h) * NN + pos) * HD + i * 2;
  d[0] = f32_to_bf16(o0);
  d[1] = f32_to_bf16(o1);
}

// ---------------------------------------------------------------- generic WMMA GEMM
// C (MxN f32) = A (MxK bf16 row-major) * Bt^T, Bt stored (N x K) bf16 row-major.
// block: 256 threads (8 waves), tile 128(M) x 64(N); wave tile 32x32.
// K-step 64, TDM double-buffered tiles, LDS row pitch 80 (64 elems + 32B TDM pad).
#define GLD 80
__global__ void __launch_bounds__(256)
gemm_bf16_wmma(const u16* __restrict__ A, const u16* __restrict__ Bt,
               float* __restrict__ C, int M, int N, int K) {
  __shared__ u16 As[2][128 * GLD];
  __shared__ u16 Bs[2][64 * GLD];
  const int tid  = threadIdx.x;
  const int lane = tid & 31;
  const int wave = tid >> 5;
  const int wm = (wave & 3) * 32;      // wave M offset in block tile
  const int wn = (wave >> 2) * 32;     // wave N offset
  const int bM = blockIdx.y * 128;
  const int bN = blockIdx.x * 64;

  f32x8 acc[2][2];
#pragma unroll
  for (int i = 0; i < 2; ++i)
#pragma unroll
    for (int j = 0; j < 2; ++j)
#pragma unroll
      for (int r = 0; r < 8; ++r) acc[i][j][r] = 0.0f;

  const int nIter = K >> 6;    // K/64
  if (tid < 32) {              // one wave drives the TDM
    tdm_load_2d(lds_off(&As[0][0]), A  + (size_t)bM * K, 64, 128, K, 4, 7);
    tdm_load_2d(lds_off(&Bs[0][0]), Bt + (size_t)bN * K, 64,  64, K, 4, 7);
  }

  for (int it = 0; it < nIter; ++it) {
    const int cur = it & 1;
    if (tid < 32) {
      if (it + 1 < nIter) {   // prefetch next tiles into the other buffer
        tdm_load_2d(lds_off(&As[1 - cur][0]), A  + (size_t)bM * K + (it + 1) * 64, 64, 128, K, 4, 7);
        tdm_load_2d(lds_off(&Bs[1 - cur][0]), Bt + (size_t)bN * K + (it + 1) * 64, 64,  64, K, 4, 7);
        __builtin_amdgcn_s_wait_tensorcnt(2);   // in-order: current tiles complete
      } else {
        __builtin_amdgcn_s_wait_tensorcnt(0);
      }
    }
    __syncthreads();

#pragma unroll
    for (int kk = 0; kk < 2; ++kk) {
      bf16x16 af[2], bf[2];
#pragma unroll
      for (int i = 0; i < 2; ++i) af[i] = frag_a(&As[cur][(wm + i * 16) * GLD + kk * 32], GLD);
#pragma unroll
      for (int j = 0; j < 2; ++j) bf[j] = frag_b(&Bs[cur][(wn + j * 16) * GLD + kk * 32], GLD);
#pragma unroll
      for (int i = 0; i < 2; ++i)
#pragma unroll
        for (int j = 0; j < 2; ++j)
          acc[i][j] = __builtin_amdgcn_wmma_f32_16x16x32_bf16(
              false, af[i], false, bf[j], (short)0, acc[i][j], false, false);
    }
    __syncthreads();   // all waves done with buf[cur] before TDM refills it
  }

#pragma unroll
  for (int i = 0; i < 2; ++i)
#pragma unroll
    for (int j = 0; j < 2; ++j) {
      int m0 = bM + wm + i * 16 + (lane >> 4) * 8;
      int n0 = bN + wn + j * 16 + (lane & 15);
      float* cp = C + (size_t)m0 * N + n0;
#pragma unroll
      for (int r = 0; r < 8; ++r) cp[(size_t)r * N] = acc[i][j][r];
    }
}

// ---------------------------------------------------------------- flash attention
// grid (NN/64, HH, BB); block 128 (4 waves). Each block: 64 queries of one head.
#define ALD 80   // 64-wide tiles: TDM pad (interval 4, amount 7) -> pitch 160B
#define SLD 48   // 32-wide tiles (manual staging)
__global__ void __launch_bounds__(128)
attn_fwd_wmma(const u16* __restrict__ Qb, const u16* __restrict__ Kb,
              const u16* __restrict__ Vb, u16* __restrict__ Ob) {
  __shared__ u16  Qs[64 * ALD];
  __shared__ u16  Ks[32 * ALD];
  __shared__ u16  Vts[64 * SLD];     // transposed: [d][j]
  __shared__ u16  Ps[64 * SLD];
  __shared__ float Ss[64 * 32];
  __shared__ float m_run[64], l_run[64], scl[64];

  const int tid = threadIdx.x, lane = tid & 31, wave = tid >> 5;
  const int i0 = blockIdx.x * 64;
  const int h  = blockIdx.y;
  const int b  = blockIdx.z;
  const int kvh = h / GG;
  const u16* Qh = Qb + ((size_t)b * HH  + h)   * NN * HD;
  const u16* Kh = Kb + ((size_t)b * KVH + kvh) * NN * HD;
  const u16* Vh = Vb + ((size_t)b * KVH + kvh) * NN * HD;

  if (tid < 32)    // Q tile 64x64 via TDM
    tdm_load_2d(lds_off(Qs), Qh + (size_t)i0 * HD, 64, 64, HD, 4, 7);
  if (tid < 64) { m_run[tid] = -1e30f; l_run[tid] = 0.0f; }

  f32x8 o[4];
#pragma unroll
  for (int c = 0; c < 4; ++c)
#pragma unroll
    for (int r = 0; r < 8; ++r) o[c][r] = 0.0f;

  const int qrow = wave * 16;          // wave's 16 query rows within tile

  for (int j0 = 0; j0 < i0 + 64; j0 += 32) {
    if (tid < 32)  // K block 32x64 via TDM
      tdm_load_2d(lds_off(Ks), Kh + (size_t)j0 * HD, 64, 32, HD, 4, 7);
    {   // V block 32x64 -> LDS transposed (TDM has no transpose)
      int jj = tid >> 2, dbase = (tid & 3) * 16;
      const u16* g = Vh + (size_t)(j0 + jj) * HD + dbase;
#pragma unroll
      for (int u = 0; u < 16; ++u) Vts[(dbase + u) * SLD + jj] = g[u];
    }
    if (tid < 32) __builtin_amdgcn_s_wait_tensorcnt(0);  // covers Q on 1st iter too
    __syncthreads();

    // S = Q * K^T for rows [qrow, qrow+16), 32 key columns (two 16-col WMMA tiles)
#pragma unroll
    for (int t2 = 0; t2 < 2; ++t2) {
      f32x8 s;
#pragma unroll
      for (int r = 0; r < 8; ++r) s[r] = 0.0f;
#pragma unroll
      for (int kk = 0; kk < 2; ++kk) {
        bf16x16 a  = frag_a(&Qs[qrow * ALD + kk * 32], ALD);
        bf16x16 bm = frag_b(&Ks[(t2 * 16) * ALD + kk * 32], ALD);
        s = __builtin_amdgcn_wmma_f32_16x16x32_bf16(false, a, false, bm, (short)0, s, false, false);
      }
      int col = t2 * 16 + (lane & 15);
      int rb  = qrow + (lane >> 4) * 8;
#pragma unroll
      for (int r = 0; r < 8; ++r) {
        float val = s[r] * 0.125f;                    // 1/sqrt(64)
        if (j0 + col > i0 + rb + r) val = -1e30f;     // causal mask
        Ss[(rb + r) * 32 + col] = val;
      }
    }
    __syncthreads();

    // online softmax row update (one thread per row, stats in LDS)
    if (tid < 64) {
      float mb = -1e30f;
      for (int c = 0; c < 32; ++c) mb = fmaxf(mb, Ss[tid * 32 + c]);
      float mo = m_run[tid];
      float mn = fmaxf(mo, mb);
      float sc = __expf(mo - mn);
      float ps = 0.0f;
      for (int c = 0; c < 32; ++c) {
        float p = __expf(Ss[tid * 32 + c] - mn);
        Ps[tid * SLD + c] = f32_to_bf16(p);
        ps += p;
      }
      m_run[tid] = mn;
      l_run[tid] = l_run[tid] * sc + ps;
      scl[tid] = sc;
    }
    __syncthreads();

    // rescale O, accumulate O += P * V (A-frag of P reused across 4 d-tiles)
    float sr[8];
#pragma unroll
    for (int r = 0; r < 8; ++r) sr[r] = scl[qrow + (lane >> 4) * 8 + r];
    bf16x16 a = frag_a(&Ps[qrow * SLD], SLD);
#pragma unroll
    for (int c = 0; c < 4; ++c) {
#pragma unroll
      for (int r = 0; r < 8; ++r) o[c][r] *= sr[r];
      bf16x16 bm = frag_b(&Vts[(c * 16) * SLD], SLD);
      o[c] = __builtin_amdgcn_wmma_f32_16x16x32_bf16(false, a, false, bm, (short)0, o[c], false, false);
    }
    __syncthreads();   // protect LDS before next iteration's loads / TDM
  }

  // normalize and write O as bf16 into (B*N, H*HD) for the out-projection GEMM
  float inv[8];
#pragma unroll
  for (int r = 0; r < 8; ++r) inv[r] = 1.0f / l_run[qrow + (lane >> 4) * 8 + r];
#pragma unroll
  for (int c = 0; c < 4; ++c) {
#pragma unroll
    for (int r = 0; r < 8; ++r) {
      int mrel = qrow + (lane >> 4) * 8 + r;
      int d    = c * 16 + (lane & 15);
      Ob[((size_t)b * NN + i0 + mrel) * (HH * HD) + h * HD + d] = f32_to_bf16(o[c][r] * inv[r]);
    }
  }
}

// ---------------------------------------------------------------- host launcher
extern "C" void kernel_launch(void* const* d_in, const int* in_sizes, int n_in,
                              void* d_out, int out_size, void* d_ws, size_t ws_size,
                              hipStream_t stream) {
  const float* x  = (const float*)d_in[0];
  const float* Wq = (const float*)d_in[1];
  const float* Wk = (const float*)d_in[2];
  const float* Wv = (const float*)d_in[3];
  const float* Wo = (const float*)d_in[4];
  const float* fc = (const float*)d_in[5];
  const float* fs = (const float*)d_in[6];
  float* out = (float*)d_out;

  char* ws = (char*)d_ws;
  size_t off = 0;
  auto alloc = [&](size_t bytes) -> void* {
    void* p = ws + off;
    off = (off + bytes + 255) & ~(size_t)255;
    return p;
  };

  u16*  xb  = (u16*)alloc((size_t)MM * DD * 2);            // x bf16
  u16*  WqT = (u16*)alloc((size_t)(HH * HD) * DD * 2);     // (2048, 2048)
  u16*  WkT = (u16*)alloc((size_t)(KVH * HD) * DD * 2);    // (512, 2048)
  u16*  WvT = (u16*)alloc((size_t)(KVH * HD) * DD * 2);
  u16*  WoT = (u16*)alloc((size_t)DD * (HH * HD) * 2);
  float* Qf = (float*)alloc((size_t)MM * HH * HD * 4);
  float* Kf = (float*)alloc((size_t)MM * KVH * HD * 4);
  float* Vf = (float*)alloc((size_t)MM * KVH * HD * 4);
  u16*  Qbf = (u16*)alloc((size_t)BB * HH * NN * HD * 2);
  u16*  Kbf = (u16*)alloc((size_t)BB * KVH * NN * HD * 2);
  u16*  Vbf = (u16*)alloc((size_t)BB * KVH * NN * HD * 2);
  u16*  Obf = (u16*)alloc((size_t)MM * HH * HD * 2);

  // 1. convert x to bf16
  {
    size_t n = (size_t)MM * DD;
    cvt_bf16_kernel<<<(unsigned)((n + 255) / 256), 256, 0, stream>>>(x, xb, n);
  }
  // 2. transpose + convert weights
  {
    size_t n = (size_t)DD * (HH * HD);
    transpose_cvt_kernel<<<(unsigned)((n + 255) / 256), 256, 0, stream>>>(Wq, WqT, DD, HH * HD);
    size_t m = (size_t)DD * (KVH * HD);
    transpose_cvt_kernel<<<(unsigned)((m + 255) / 256), 256, 0, stream>>>(Wk, WkT, DD, KVH * HD);
    transpose_cvt_kernel<<<(unsigned)((m + 255) / 256), 256, 0, stream>>>(Wv, WvT, DD, KVH * HD);
    transpose_cvt_kernel<<<(unsigned)((n + 255) / 256), 256, 0, stream>>>(Wo, WoT, HH * HD, DD);
  }
  // 3. QKV projection GEMMs
  {
    dim3 blk(256);
    gemm_bf16_wmma<<<dim3((HH * HD) / 64, MM / 128), blk, 0, stream>>>(xb, WqT, Qf, MM, HH * HD, DD);
    gemm_bf16_wmma<<<dim3((KVH * HD) / 64, MM / 128), blk, 0, stream>>>(xb, WkT, Kf, MM, KVH * HD, DD);
    gemm_bf16_wmma<<<dim3((KVH * HD) / 64, MM / 128), blk, 0, stream>>>(xb, WvT, Vf, MM, KVH * HD, DD);
  }
  // 4. RoPE + permute to (B, heads, N, HD) bf16
  {
    size_t nq = (size_t)BB * HH * NN * (HD / 2);
    rope_cvt_kernel<<<(unsigned)((nq + 255) / 256), 256, 0, stream>>>(Qf, Qbf, fc, fs, HH, 1);
    size_t nk = (size_t)BB * KVH * NN * (HD / 2);
    rope_cvt_kernel<<<(unsigned)((nk + 255) / 256), 256, 0, stream>>>(Kf, Kbf, fc, fs, KVH, 1);
    rope_cvt_kernel<<<(unsigned)((nk + 255) / 256), 256, 0, stream>>>(Vf, Vbf, fc, fs, KVH, 0);
  }
  // 5. flash attention
  attn_fwd_wmma<<<dim3(NN / 64, HH, BB), 128, 0, stream>>>(Qbf, Kbf, Vbf, Obf);
  // 6. output projection
  gemm_bf16_wmma<<<dim3(DD / 64, MM / 128), 256, 0, stream>>>(Obf, WoT, out, MM, DD, HH * HD);
}
